// Stem_slic_V2_1_23845658427411
// MI455X (gfx1250) — matrix operand order repaired and checked by reading, not compiled, not run on hardware
//
#include <hip/hip_runtime.h>

typedef __attribute__((ext_vector_type(16))) _Float16 v16h;
typedef __attribute__((ext_vector_type(8)))  float    v8f;
typedef __attribute__((ext_vector_type(4)))  float    f4raw;

union Frag { v16h h; f4raw q[2]; };

#define TPB 256

// ---------------- workspace layout (bytes, all 256-aligned) ----------------
#define OFF_WP    0u                       // packed weights f16   : 5,308,416 elems
#define OFF_COL   10616832u                // im2col f16           : 10,838,016 elems
#define OFF_ACTA  32292864u                // act ping  f32        : 1,572,864 elems
#define OFF_ACTB  38584320u                // act pong  f32        : 1,572,864 elems
#define OFF_STATS 44875776u                // per-channel sum/sumsq: 2*768 f32
#define OFF_WINT  44881920u                // mask counts i32      : 8*196*196
#define OFF_WT    46111232u                // mask f16 [8][208][224]
#define OFF_CNT   46856704u                // counts f32 [8][208]
#define OFF_FEATH 46863360u                // feat f16 [8][768][224]

// d_out float offsets: segmaps | pooled out | feat
#define SEG_N   401408
#define OUT_OFF 401408
#define FEAT_OFF 1605632

// ---------------------------------------------------------------- utilities
__global__ void k_zero_i32(int* __restrict__ p, int n) {
  int i = blockIdx.x * TPB + threadIdx.x;
  if (i < n) p[i] = 0;
}

__global__ void k_seg_copy(const int* __restrict__ seg, float* __restrict__ out) {
  int i = blockIdx.x * TPB + threadIdx.x;
  if (i < SEG_N) out[i] = (float)seg[i];
}

// ------------------------------------------------- superpixel count matrix
// mr(b,m,y,x)=1 iff any of segmaps[b, 4y+1..4y+2, 4x+1..4x+2]==m  (exact
// reduction of bilinear 224->56, w=0.5, thresholded >0.1).
// K(b,m,p) = #(y,x) in the 4x4 block of superpatch p where mr==1.
__global__ void k_build_mask(const int* __restrict__ seg, int* __restrict__ W) {
  int idx = blockIdx.x * TPB + threadIdx.x;
  if (idx >= 8 * 56 * 56) return;
  int b = idx / 3136, rem = idx % 3136, y = rem / 56, x = rem % 56;
  const int* sb = seg + b * 224 * 224;
  int r0 = 4 * y + 1, c0 = 4 * x + 1;
  int s0 = sb[r0 * 224 + c0],       s1 = sb[r0 * 224 + c0 + 1];
  int s2 = sb[(r0 + 1) * 224 + c0], s3 = sb[(r0 + 1) * 224 + c0 + 1];
  int p = (y >> 2) * 14 + (x >> 2);
  int* wb = W + b * 196 * 196 + p;                 // [b][m][p]
  atomicAdd(wb + s0 * 196, 1);
  if (s1 != s0)                            atomicAdd(wb + s1 * 196, 1);
  if (s2 != s0 && s2 != s1)                atomicAdd(wb + s2 * 196, 1);
  if (s3 != s0 && s3 != s1 && s3 != s2)    atomicAdd(wb + s3 * 196, 1);
}

// counts -> f16 [8][208][224] (zero padded) + per-segment totals
__global__ void k_pack_cnt(const int* __restrict__ W, _Float16* __restrict__ Wt,
                           float* __restrict__ cnt) {
  int idx = blockIdx.x * TPB + threadIdx.x;
  if (idx >= 8 * 208) return;
  int b = idx / 208, m = idx % 208;
  _Float16* dst = Wt + (size_t)idx * 224;
  float s = 0.f;
  if (m < 196) {
    const int* src = W + (size_t)(b * 196 + m) * 196;
    for (int p = 0; p < 196; ++p) { int v = src[p]; s += (float)v; dst[p] = (_Float16)(float)v; }
    for (int p = 196; p < 224; ++p) dst[p] = (_Float16)0.f;
  } else {
    for (int p = 0; p < 224; ++p) dst[p] = (_Float16)0.f;
  }
  cnt[idx] = s;
}

// ---------------------------------------------------------- conv as GEMM
// weights (Co,Ci,3,3) fp32 -> f16 [Co][Kp], K=Ci*9, zero-pad K..Kp
__global__ void k_packw(const float* __restrict__ w, _Float16* __restrict__ wp,
                        int K, int Kp, int Co) {
  long idx = (long)blockIdx.x * TPB + threadIdx.x;
  if (idx >= (long)Co * Kp) return;
  int kp = (int)(idx % Kp);
  long co = idx / Kp;
  wp[idx] = (_Float16)(kp < K ? w[co * K + kp] : 0.f);
}

// im2col patches -> f16 [N][Kp]  (N = b*Ho*Wo + oh*Wo + ow, patch-major so it
// feeds the WMMA B operand with contiguous 16B loads)
__global__ void k_im2col(const float* __restrict__ in, _Float16* __restrict__ out,
                         int Ci, int Hi, int Wi, int Ho, int Wo, int stride, int pad,
                         int Kp, long strideC, long strideB, int Ntot) {
  long idx = (long)blockIdx.x * TPB + threadIdx.x;
  if (idx >= (long)Ntot * Kp) return;
  int kp = (int)(idx % Kp);
  long n = idx / Kp;
  int K = Ci * 9;
  float v = 0.f;
  if (kp < K) {
    int ci = kp / 9, r = kp % 9, ky = r / 3, kx = r % 3;
    int hw = Ho * Wo;
    int b = (int)(n / hw), rem = (int)(n % hw), oh = rem / Wo, ow = rem % Wo;
    int ih = oh * stride - pad + ky, iw = ow * stride - pad + kx;
    if (ih >= 0 && ih < Hi && iw >= 0 && iw < Wi)
      v = in[ci * strideC + b * strideB + (long)ih * Wi + iw];
  }
  out[idx] = (_Float16)v;
}

// C[M][N] = A[M][Kp] * Bt[N][Kp]^T, f16 in / f32 accumulate via WMMA.
// One 16x16 tile per wave; all guards wave-uniform (EXEC all-ones for WMMA).
__global__ void k_gemm_wmma(const _Float16* __restrict__ A,
                            const _Float16* __restrict__ Bt,
                            float* __restrict__ C,
                            int Nt, int N, int Kp, int totalTiles) {
  int wave = (int)((blockIdx.x * (unsigned)blockDim.x + threadIdx.x) >> 5);
  if (wave >= totalTiles) return;
  int lane = threadIdx.x & 31, l15 = lane & 15, hi = lane >> 4;
  int mt = wave / Nt, nt = wave % Nt;
  const _Float16* arow = A  + (size_t)(mt * 16 + l15) * Kp + hi * 8;   // A: K halves {h8.., 16+h8..}
  const _Float16* brow = Bt + (size_t)(nt * 16 + l15) * Kp + hi * 8;   // B: K halves {h8.., 16+h8..}
  v8f acc = {};
  for (int k = 0; k < Kp; k += 32) {
    Frag a, b;
    a.q[0] = *(const f4raw*)(arow + k);
    a.q[1] = *(const f4raw*)(arow + k + 16);
    b.q[0] = *(const f4raw*)(brow + k);
    b.q[1] = *(const f4raw*)(brow + k + 16);
    acc = __builtin_amdgcn_wmma_f32_16x16x32_f16(false, a.h, false, b.h,
                                                 (short)0, acc, false, false);
  }
  int rbase = hi << 3;
  float* cp = C + (size_t)(mt * 16 + rbase) * N + nt * 16 + l15;
  #pragma unroll
  for (int r = 0; r < 8; ++r) cp[(size_t)r * N] = acc[r];
}

// --------------------------------------------------------------- batchnorm
__global__ void k_bn_stats(const float* __restrict__ X, float* __restrict__ st, int N) {
  int c = blockIdx.x;
  const float* row = X + (size_t)c * N;
  float s = 0.f, ss = 0.f;
  for (int i = threadIdx.x; i < N; i += TPB) { float v = row[i]; s += v; ss += v * v; }
  __shared__ float sh0[TPB], sh1[TPB];
  sh0[threadIdx.x] = s; sh1[threadIdx.x] = ss;
  __syncthreads();
  for (int off = TPB / 2; off > 0; off >>= 1) {
    if ((int)threadIdx.x < off) { sh0[threadIdx.x] += sh0[threadIdx.x + off];
                                  sh1[threadIdx.x] += sh1[threadIdx.x + off]; }
    __syncthreads();
  }
  if (threadIdx.x == 0) { st[2 * c] = sh0[0]; st[2 * c + 1] = sh1[0]; }
}

__global__ void k_bn_apply(float* __restrict__ X, const float* __restrict__ st,
                           const float* __restrict__ g, const float* __restrict__ be,
                           const float* __restrict__ pos, int N, int C, int relu) {
  long i = (long)blockIdx.x * TPB + threadIdx.x;
  if (i >= (long)C * N) return;
  int c = (int)(i / N), n = (int)(i % N);
  float invN = 1.f / (float)N;
  float mu = st[2 * c] * invN;
  float var = st[2 * c + 1] * invN - mu * mu;            // biased, training mode
  float v = (X[i] - mu) * rsqrtf(var + 1e-5f) * g[c] + be[c];
  if (relu) v = fmaxf(v, 0.f);
  if (pos) v += pos[c * 196 + (n % 196)];
  X[i] = v;
}

// feat [768][8*196] channel-major -> d_out NCHW fp32 + f16 [8][768][224] (padded)
__global__ void k_feat_split(const float* __restrict__ act5, float* __restrict__ fo,
                             _Float16* __restrict__ fh) {
  long idx = (long)blockIdx.x * TPB + threadIdx.x;
  if (idx >= 8L * 768 * 224) return;
  int pp = (int)(idx % 224);
  long bf = idx / 224;
  int f = (int)(bf % 768), b = (int)(bf / 768);
  float v = 0.f;
  if (pp < 196) {
    v = act5[(size_t)f * 1568 + b * 196 + pp];
    fo[bf * 196 + pp] = v;
  }
  fh[idx] = (_Float16)v;
}

// pooled out(b,f,m) = (feat_b[768x224] * Wt_b[208x224]^T)/cnt via WMMA
__global__ void k_pool_gemm(const _Float16* __restrict__ fh, const _Float16* __restrict__ Wt,
                            const float* __restrict__ cnt, float* __restrict__ out) {
  const int Mt = 48, Nt = 13, tilesPerB = Mt * Nt;
  int wave = (int)((blockIdx.x * (unsigned)blockDim.x + threadIdx.x) >> 5);
  if (wave >= 8 * tilesPerB) return;
  int lane = threadIdx.x & 31, l15 = lane & 15, hi = lane >> 4;
  int b = wave / tilesPerB, t = wave % tilesPerB;
  int mt = t / Nt, nt = t % Nt;
  const _Float16* arow = fh + ((size_t)b * 768 + mt * 16 + l15) * 224 + hi * 8;
  const _Float16* brow = Wt + ((size_t)b * 208 + nt * 16 + l15) * 224 + hi * 8;
  v8f acc = {};
  #pragma unroll
  for (int k = 0; k < 224; k += 32) {
    Frag a, bb;
    a.q[0]  = *(const f4raw*)(arow + k);
    a.q[1]  = *(const f4raw*)(arow + k + 16);
    bb.q[0] = *(const f4raw*)(brow + k);
    bb.q[1] = *(const f4raw*)(brow + k + 16);
    acc = __builtin_amdgcn_wmma_f32_16x16x32_f16(false, a.h, false, bb.h,
                                                 (short)0, acc, false, false);
  }
  int col = nt * 16 + l15;
  if (col < 196) {                       // after all WMMAs; divergence OK here
    float c = cnt[b * 208 + col];
    float inv = (c > 0.f) ? 1.f / c : 0.f;   // cnt==0 -> num==0 -> nan_to_num==0
    int rbase = hi << 3;
    float* op = out + ((size_t)b * 768 + mt * 16 + rbase) * 196 + col;
    #pragma unroll
    for (int r = 0; r < 8; ++r) op[(size_t)r * 196] = acc[r] * inv;
  }
}

// ------------------------------------------------------------------- host
extern "C" void kernel_launch(void* const* d_in, const int* in_sizes, int n_in,
                              void* d_out, int out_size, void* d_ws, size_t ws_size,
                              hipStream_t stream) {
  const float* x   = (const float*)d_in[0];
  const int*   seg = (const int*)d_in[1];
  const float* pos = (const float*)d_in[22];
  float* fout = (float*)d_out;
  char*  ws   = (char*)d_ws;

  _Float16* wp    = (_Float16*)(ws + OFF_WP);
  _Float16* col   = (_Float16*)(ws + OFF_COL);
  float*    actA  = (float*)(ws + OFF_ACTA);
  float*    actB  = (float*)(ws + OFF_ACTB);
  float*    stats = (float*)(ws + OFF_STATS);
  int*      wint  = (int*)(ws + OFF_WINT);
  _Float16* wt    = (_Float16*)(ws + OFF_WT);
  float*    cnt   = (float*)(ws + OFF_CNT);
  _Float16* feath = (_Float16*)(ws + OFF_FEATH);

  // ---- segment count matrix + segmaps passthrough
  {
    int n = 8 * 196 * 196;
    k_zero_i32<<<(n + TPB - 1) / TPB, TPB, 0, stream>>>(wint, n);
    k_build_mask<<<(8 * 56 * 56 + TPB - 1) / TPB, TPB, 0, stream>>>(seg, wint);
    k_pack_cnt<<<(8 * 208 + TPB - 1) / TPB, TPB, 0, stream>>>(wint, wt, cnt);
    k_seg_copy<<<(SEG_N + TPB - 1) / TPB, TPB, 0, stream>>>(seg, fout);
  }

  // ---- stem: conv(im2col+WMMA) -> BN(batch stats) -> relu, in [C][B*H*W]
  struct LCfg { int Ci, Hi, Wi, Ho, Wo, s, p, Co, relu; };
  const LCfg L[5] = {
    {   3, 64, 64, 32, 32, 2, 1,  96, 1 },
    {  96, 32, 32, 16, 16, 2, 1, 192, 1 },
    { 192, 16, 16, 16, 16, 1, 1, 384, 1 },
    { 384, 16, 16, 16, 16, 1, 1, 768, 1 },
    { 768, 16, 16, 14, 14, 1, 0, 768, 0 },
  };
  float* acts[5] = { actA, actB, actA, actB, actA };
  const float* in = x;
  long sc = 64 * 64, sb = 3 * 64 * 64;          // input x is NCHW
  for (int l = 0; l < 5; ++l) {
    const LCfg& c = L[l];
    int K = c.Ci * 9, Kp = (K + 31) & ~31;
    int N = 8 * c.Ho * c.Wo;
    const float* w  = (const float*)d_in[2 + 4 * l];
    const float* g  = (const float*)d_in[4 + 4 * l];
    const float* be = (const float*)d_in[5 + 4 * l];
    float* out = acts[l];

    long nwp = (long)c.Co * Kp;
    k_packw<<<(unsigned)((nwp + TPB - 1) / TPB), TPB, 0, stream>>>(w, wp, K, Kp, c.Co);
    long ncol = (long)N * Kp;
    k_im2col<<<(unsigned)((ncol + TPB - 1) / TPB), TPB, 0, stream>>>(
        in, col, c.Ci, c.Hi, c.Wi, c.Ho, c.Wo, c.s, c.p, Kp, sc, sb, N);
    int Nt = N / 16, tiles = (c.Co / 16) * Nt;
    k_gemm_wmma<<<(tiles + 7) / 8, TPB, 0, stream>>>(wp, col, out, Nt, N, Kp, tiles);
    k_bn_stats<<<c.Co, TPB, 0, stream>>>(out, stats, N);
    long napp = (long)c.Co * N;
    k_bn_apply<<<(unsigned)((napp + TPB - 1) / TPB), TPB, 0, stream>>>(
        out, stats, g, be, (l == 4) ? pos : (const float*)nullptr, N, c.Co, c.relu);
    in = out; sc = N; sb = c.Ho * c.Wo;         // activations are [C][B*H*W]
  }

  // ---- feat outputs + pooled GEMM
  k_feat_split<<<(unsigned)((8L * 768 * 224 + TPB - 1) / TPB), TPB, 0, stream>>>(
      acts[4], fout + FEAT_OFF, feath);
  int poolWaves = 8 * 48 * 13;
  k_pool_gemm<<<(poolWaves + 7) / 8, TPB, 0, stream>>>(feath, wt, cnt, fout + OUT_OFF);
}
